// EncoderLayer_78821239816781
// MI455X (gfx1250) — matrix operand order repaired and checked by reading, not compiled, not run on hardware
//
#include <hip/hip_runtime.h>
#include <hip/hip_bf16.h>

// ---------------------------------------------------------------------------
// Autoformer-style encoder layer for MI455X (gfx1250, wave32, WMMA).
// All matmuls use v_wmma_f32_16x16x32_f16 (f16 in, f32 accumulate).
// - GEMMs use 2x2 register blocking (32x32 tile / wave): each A/B fragment
//   feeds two WMMAs -> 16 flops/byte of VMEM instead of 8.
// - attn [B,H,L,L] (537 MB) is written exactly once: the attention kernel
//   keeps a 16x2048 f32 score stripe in LDS (WGP has 320KB), does softmax
//   there, streams attn out while mirroring probs to an f16 LDS buffer so
//   the attn@V A-fragments are two ds_load_b128 per lane (not 16 scalars).
// ---------------------------------------------------------------------------

typedef __attribute__((ext_vector_type(16))) _Float16 v16h;
typedef __attribute__((ext_vector_type(8)))  _Float16 v8h;
typedef __attribute__((ext_vector_type(8)))  float    v8f;

#define WMMA_F16(a, b, c) \
  __builtin_amdgcn_wmma_f32_16x16x32_f16(false, (a), false, (b), (short)0, (c), false, false)

// Build a 16-bit A-fragment in contiguous K order (B loads use the same):
// lane&15 = row M; kb16 = (lane>>4)*16; elem j -> K=kb16+j
__device__ inline v16h ld_a16(const _Float16* __restrict__ p0,
                              const _Float16* __restrict__ p1) {
  v8h lo = *(const v8h*)p0;
  v8h hi = *(const v8h*)p1;
  v16h r;
#pragma unroll
  for (int j = 0; j < 8; ++j) { r[j] = lo[j]; r[j + 8] = hi[j]; }
  return r;
}

// ---------------------------------------------------------------------------
// Elementwise prep kernels
// ---------------------------------------------------------------------------
__global__ void k_cvt_f32_f16(const float* __restrict__ src,
                              _Float16* __restrict__ dst, int n) {
  int i = blockIdx.x * blockDim.x + threadIdx.x;
  if (i < n) dst[i] = (_Float16)src[i];
}

// dst[c*rows + r] = src[r*cols + c]  (gives K-contiguous BT for x @ W)
__global__ void k_transpose_cvt(const float* __restrict__ src,
                                _Float16* __restrict__ dst, int rows, int cols) {
  int i = blockIdx.x * blockDim.x + threadIdx.x;
  if (i >= rows * cols) return;
  int r = i / cols, c = i % cols;
  dst[c * rows + r] = (_Float16)src[i];
}

// Circular moving average (width 7) decomposition along L=2048 (power of 2).
// out32 = in - mean7(in);  out16 optional f16 copy for subsequent WMMA GEMM.
__global__ void k_decomp(const float* __restrict__ in,
                         float* __restrict__ out32,
                         _Float16* __restrict__ out16, int total) {
  int idx = blockIdx.x * blockDim.x + threadIdx.x;
  if (idx >= total) return;
  int c = idx & 255;
  int l = (idx >> 8) & 2047;
  int bbase = idx & ~(2048 * 256 - 1);
  float s = 0.f;
#pragma unroll
  for (int off = -3; off <= 3; ++off) {
    int ll = (l + off + 2048) & 2047;
    s += in[bbase + (ll << 8) + c];
  }
  float v = in[idx] - s * (1.0f / 7.0f);
  out32[idx] = v;
  if (out16) out16[idx] = (_Float16)v;
}

// ---------------------------------------------------------------------------
// C-fragment store helpers (16x16 f32 C layout: M = (lane>>4)*8 + j, N = lane&15)
// ---------------------------------------------------------------------------
template <bool RELU, bool VTRANS>
__device__ inline void store_c16(_Float16* __restrict__ C, const v8f& acc,
                                 int mbase, int nbase, int lane, int N) {
  const int nc = nbase + (lane & 15);
#pragma unroll
  for (int j = 0; j < 8; ++j) {
    int mr = mbase + (lane >> 4) * 8 + j;
    float v = acc[j];
    if (RELU && v < 0.f) v = 0.f;
    if (VTRANS) {
      int b_ = mr >> 11, l_ = mr & 2047;               // N == 256 here
      C[((size_t)(b_ * 256 + nc)) * 2048 + l_] = (_Float16)v;
    } else {
      C[(size_t)mr * N + nc] = (_Float16)v;
    }
  }
}

__device__ inline void store_c32res(float* __restrict__ C,
                                    const float* __restrict__ R, const v8f& acc,
                                    int mbase, int nbase, int lane, int N) {
  const int nc = nbase + (lane & 15);
#pragma unroll
  for (int j = 0; j < 8; ++j) {
    size_t idx = (size_t)(mbase + (lane >> 4) * 8 + j) * N + nc;
    C[idx] = acc[j] + R[idx];
  }
}

// ---------------------------------------------------------------------------
// WMMA GEMM, 2x2 register blocked: one wave computes a 32x32 output tile.
// A[M,K] f16 row-major; BT stored [N,K] f16 (K-contiguous).
// ---------------------------------------------------------------------------
template <bool RELU, bool VTRANS>
__global__ __launch_bounds__(32) void k_gemm32(const _Float16* __restrict__ A,
                                               const _Float16* __restrict__ BT,
                                               _Float16* __restrict__ C,
                                               int M, int N, int K) {
  const int lane = threadIdx.x & 31;
  const int m0 = blockIdx.x * 32;
  const int n0 = blockIdx.y * 32;
  const int kb   = (lane >> 4) * 8;
  const int kb16 = (lane >> 4) * 16;
  const _Float16* __restrict__ a0 = A + (size_t)(m0 + (lane & 15)) * K;
  const _Float16* __restrict__ a1 = a0 + (size_t)16 * K;
  const _Float16* __restrict__ b0 = BT + (size_t)(n0 + (lane & 15)) * K;
  const _Float16* __restrict__ b1 = b0 + (size_t)16 * K;
  v8f c00 = {}, c01 = {}, c10 = {}, c11 = {};
#pragma unroll 2
  for (int kk = 0; kk < K; kk += 32) {
    v16h A0 = ld_a16(a0 + kk + kb16, a0 + kk + kb16 + 8);
    v16h A1 = ld_a16(a1 + kk + kb16, a1 + kk + kb16 + 8);
    v16h B0 = *(const v16h*)(b0 + kk + kb16);
    v16h B1 = *(const v16h*)(b1 + kk + kb16);
    c00 = WMMA_F16(A0, B0, c00);
    c01 = WMMA_F16(A0, B1, c01);
    c10 = WMMA_F16(A1, B0, c10);
    c11 = WMMA_F16(A1, B1, c11);
  }
  store_c16<RELU, VTRANS>(C, c00, m0,      n0,      lane, N);
  store_c16<RELU, VTRANS>(C, c01, m0,      n0 + 16, lane, N);
  store_c16<RELU, VTRANS>(C, c10, m0 + 16, n0,      lane, N);
  store_c16<RELU, VTRANS>(C, c11, m0 + 16, n0 + 16, lane, N);
}

// f32 output + residual add: C = A x BT + R   (Wo projection, FFN2)
__global__ __launch_bounds__(32) void k_gemm32_f32res(const _Float16* __restrict__ A,
                                                      const _Float16* __restrict__ BT,
                                                      const float* __restrict__ R,
                                                      float* __restrict__ C,
                                                      int M, int N, int K) {
  const int lane = threadIdx.x & 31;
  const int m0 = blockIdx.x * 32;
  const int n0 = blockIdx.y * 32;
  const int kb   = (lane >> 4) * 8;
  const int kb16 = (lane >> 4) * 16;
  const _Float16* __restrict__ a0 = A + (size_t)(m0 + (lane & 15)) * K;
  const _Float16* __restrict__ a1 = a0 + (size_t)16 * K;
  const _Float16* __restrict__ b0 = BT + (size_t)(n0 + (lane & 15)) * K;
  const _Float16* __restrict__ b1 = b0 + (size_t)16 * K;
  v8f c00 = {}, c01 = {}, c10 = {}, c11 = {};
#pragma unroll 2
  for (int kk = 0; kk < K; kk += 32) {
    v16h A0 = ld_a16(a0 + kk + kb16, a0 + kk + kb16 + 8);
    v16h A1 = ld_a16(a1 + kk + kb16, a1 + kk + kb16 + 8);
    v16h B0 = *(const v16h*)(b0 + kk + kb16);
    v16h B1 = *(const v16h*)(b1 + kk + kb16);
    c00 = WMMA_F16(A0, B0, c00);
    c01 = WMMA_F16(A0, B1, c01);
    c10 = WMMA_F16(A1, B0, c10);
    c11 = WMMA_F16(A1, B1, c11);
  }
  store_c32res(C, R, c00, m0,      n0,      lane, N);
  store_c32res(C, R, c01, m0,      n0 + 16, lane, N);
  store_c32res(C, R, c10, m0 + 16, n0,      lane, N);
  store_c32res(C, R, c11, m0 + 16, n0 + 16, lane, N);
}

// ---------------------------------------------------------------------------
// Fused attention: per (b, h, 16-row l-tile). 128 threads = 4 waves.
// LDS: S (16x2048 f32 scores) + Sh (16x2048 f16 probs) + reduce scratch.
// Phase 1: scores via WMMA (Dh=32 -> one 16x16x32 per tile; q fragment held
//          in registers across all 32 WMMAs of the wave; 2 tiles/iter).
// Phase 2: row max / row sum.
// Phase 3: normalize, write attn to d_out, mirror probs as f16 into Sh.
// Phase 4: ctx = attn @ V via WMMA; A-fragments = 2 x ds_load_b128 from Sh.
// ---------------------------------------------------------------------------
__global__ __launch_bounds__(128) void k_attn(const _Float16* __restrict__ Qh,
                                              const _Float16* __restrict__ Kh,
                                              const _Float16* __restrict__ Vt,
                                              _Float16* __restrict__ Ctx,
                                              float* __restrict__ attn) {
  extern __shared__ __align__(16) char smem_raw[];
  float*    S    = (float*)smem_raw;                       // 16*2048 f32
  _Float16* Sh   = (_Float16*)(smem_raw + 131072);         // 16*2048 f16
  float*    rmax = (float*)(smem_raw + 131072 + 65536);    // 16
  float*    rsum = rmax + 16;                              // 16
  float*    ctxp = rsum + 16;                              // 4*16*32 (+scratch)

  const int tid  = threadIdx.x;
  const int wave = tid >> 5;
  const int lane = tid & 31;
  const int b = blockIdx.z, h = blockIdx.y;
  const int l0 = blockIdx.x * 16;
  const int kb   = (lane >> 4) * 8;
  const int kb16 = (lane >> 4) * 16;

  // ---- Phase 1: scores (q fragment reused across the whole row stripe) ----
  const _Float16* qrow =
      Qh + ((size_t)(b * 2048 + l0 + (lane & 15))) * 256 + h * 32;
  v16h aq = ld_a16(qrow + kb16, qrow + kb16 + 8);
  const float scale = 0.17677669529663687f;  // 1/sqrt(32)
  const _Float16* kbase =
      Kh + ((size_t)(b * 2048 + (lane & 15))) * 256 + h * 32 + kb16;
  for (int tp = wave; tp < 64; tp += 4) {
    int mm0 = tp * 32;
    const _Float16* k0 = kbase + (size_t)mm0 * 256;
    __builtin_prefetch(k0 + (size_t)128 * 256, 0, 1);  // next tile-pair for this wave
    v16h bk0 = *(const v16h*)k0;
    v16h bk1 = *(const v16h*)(k0 + (size_t)16 * 256);
    v8f c0 = {}, c1 = {};
    c0 = WMMA_F16(aq, bk0, c0);
    c1 = WMMA_F16(aq, bk1, c1);
#pragma unroll
    for (int j = 0; j < 8; ++j) {
      int Mr = (lane >> 4) * 8 + j;
      S[Mr * 2048 + mm0 + (lane & 15)]      = c0[j] * scale;
      S[Mr * 2048 + mm0 + 16 + (lane & 15)] = c1[j] * scale;
    }
  }
  __syncthreads();

  // ---- Phase 2: row max / row sum (8 threads per row) ----
  {
    int r = tid >> 3, seg = tid & 7;
    float mx = -1e30f;
    for (int m = seg * 256; m < seg * 256 + 256; ++m)
      mx = fmaxf(mx, S[r * 2048 + m]);
    ctxp[r * 8 + seg] = mx;
    __syncthreads();
    if (tid < 16) {
      float m2 = -1e30f;
      for (int s = 0; s < 8; ++s) m2 = fmaxf(m2, ctxp[tid * 8 + s]);
      rmax[tid] = m2;
    }
    __syncthreads();
    float rm = rmax[r], sm = 0.f;
    for (int m = seg * 256; m < seg * 256 + 256; ++m)
      sm += __expf(S[r * 2048 + m] - rm);
    ctxp[r * 8 + seg] = sm;
    __syncthreads();
    if (tid < 16) {
      float s2 = 0.f;
      for (int s = 0; s < 8; ++s) s2 += ctxp[tid * 8 + s];
      rsum[tid] = s2;
    }
    __syncthreads();
  }

  // ---- Phase 3: normalize, write attn, mirror f16 probs into Sh ----
  for (int idx = tid; idx < 16 * 2048; idx += 128) {
    int r = idx >> 11, m = idx & 2047;
    float p = __expf(S[idx] - rmax[r]) / rsum[r];
    Sh[idx] = (_Float16)p;
    attn[((size_t)((b * 8 + h) * 2048 + l0 + r)) * 2048 + m] = p;
  }
  __syncthreads();

  // ---- Phase 4: ctx = attn @ V (A from Sh via 2x ds_load_b128) ----
  const int dcol = lane & 15;
  const _Float16* shrow = Sh + (lane & 15) * 2048;
  const _Float16* vb0 = Vt + ((size_t)(b * 256 + h * 32 + dcol)) * 2048;
  const _Float16* vb1 = Vt + ((size_t)(b * 256 + h * 32 + 16 + dcol)) * 2048;
  v8f acc0 = {}, acc1 = {};
  for (int t = wave; t < 128; t += 4) {
    int mm0 = t * 16;
    v16h ap = ld_a16(shrow + mm0 + kb16, shrow + mm0 + kb16 + 8);
    v16h b0 = *(const v16h*)(vb0 + mm0 + kb16);
    v16h b1 = *(const v16h*)(vb1 + mm0 + kb16);
    acc0 = WMMA_F16(ap, b0, acc0);
    acc1 = WMMA_F16(ap, b1, acc1);
  }
#pragma unroll
  for (int j = 0; j < 8; ++j) {
    int Mr = (lane >> 4) * 8 + j;
    ctxp[wave * 512 + Mr * 32 + dcol]      = acc0[j];
    ctxp[wave * 512 + Mr * 32 + 16 + dcol] = acc1[j];
  }
  __syncthreads();
  for (int idx = tid; idx < 512; idx += 128) {
    int r = idx >> 5, d = idx & 31;
    float s = ctxp[idx] + ctxp[512 + idx] + ctxp[1024 + idx] + ctxp[1536 + idx];
    Ctx[((size_t)(b * 2048 + l0 + r)) * 256 + h * 32 + d] = (_Float16)s;
  }
}

// ---------------------------------------------------------------------------
// Host launcher
// ---------------------------------------------------------------------------
extern "C" void kernel_launch(void* const* d_in, const int* in_sizes, int n_in,
                              void* d_out, int out_size, void* d_ws, size_t ws_size,
                              hipStream_t stream) {
  const float* x  = (const float*)d_in[0];  // [4,2048,256]
  const float* Wq = (const float*)d_in[1];
  const float* Wk = (const float*)d_in[2];
  const float* Wv = (const float*)d_in[3];
  const float* Wo = (const float*)d_in[4];
  const float* W1 = (const float*)d_in[5];  // [1024,256]
  const float* W2 = (const float*)d_in[6];  // [256,1024]

  const int M = 8192, D = 256, Dff = 1024, XN = M * D;  // XN = 2097152
  float* res_out  = (float*)d_out;                      // [8192,256]
  float* attn_out = (float*)d_out + (size_t)XN;         // [4,8,2048,2048]

  char* ws = (char*)d_ws;
  size_t off = 0;
  auto alloc = [&](size_t bytes) -> void* {
    void* p = ws + off;
    off = (off + bytes + 255) & ~(size_t)255;
    return p;
  };
  _Float16* Xh  = (_Float16*)alloc((size_t)XN * 2);
  _Float16* WqT = (_Float16*)alloc((size_t)D * D * 2);
  _Float16* WkT = (_Float16*)alloc((size_t)D * D * 2);
  _Float16* WvT = (_Float16*)alloc((size_t)D * D * 2);
  _Float16* WoT = (_Float16*)alloc((size_t)D * D * 2);
  _Float16* W1h = (_Float16*)alloc((size_t)Dff * D * 2);
  _Float16* W2h = (_Float16*)alloc((size_t)D * Dff * 2);
  _Float16* Qh  = (_Float16*)alloc((size_t)XN * 2);
  _Float16* Kh  = (_Float16*)alloc((size_t)XN * 2);
  _Float16* Vt  = (_Float16*)alloc((size_t)XN * 2);
  _Float16* Ctx = (_Float16*)alloc((size_t)XN * 2);
  float*    xr  = (float*)alloc((size_t)XN * 4);
  float*    x1  = (float*)alloc((size_t)XN * 4);
  _Float16* x1h = (_Float16*)alloc((size_t)XN * 2);
  _Float16* Hh  = (_Float16*)alloc((size_t)M * Dff * 2);
  float*    z   = (float*)alloc((size_t)XN * 4);

  // --- prep: f16 conversions + weight transposes ---
  k_cvt_f32_f16<<<(XN + 255) / 256, 256, 0, stream>>>(x, Xh, XN);
  k_transpose_cvt<<<(D * D + 255) / 256, 256, 0, stream>>>(Wq, WqT, D, D);
  k_transpose_cvt<<<(D * D + 255) / 256, 256, 0, stream>>>(Wk, WkT, D, D);
  k_transpose_cvt<<<(D * D + 255) / 256, 256, 0, stream>>>(Wv, WvT, D, D);
  k_transpose_cvt<<<(D * D + 255) / 256, 256, 0, stream>>>(Wo, WoT, D, D);
  k_cvt_f32_f16<<<(Dff * D + 255) / 256, 256, 0, stream>>>(W1, W1h, Dff * D);
  k_cvt_f32_f16<<<(D * Dff + 255) / 256, 256, 0, stream>>>(W2, W2h, D * Dff);

  // --- QKV projections (WMMA, 32x32 tiles/wave) ---
  dim3 g1(M / 32, D / 32);
  k_gemm32<false, false><<<g1, 32, 0, stream>>>(Xh, WqT, Qh, M, D, D);
  k_gemm32<false, false><<<g1, 32, 0, stream>>>(Xh, WkT, Kh, M, D, D);
  k_gemm32<false, true ><<<g1, 32, 0, stream>>>(Xh, WvT, Vt, M, D, D);

  // --- fused attention (scores + softmax + attn write + attn@V) ---
  const int SMEM = 131072 + 65536 + (16 + 16 + 4 * 16 * 32) * (int)sizeof(float);
  (void)hipFuncSetAttribute(reinterpret_cast<const void*>(k_attn),
                            hipFuncAttributeMaxDynamicSharedMemorySize, SMEM);
  dim3 ga(2048 / 16, 8, 4);
  k_attn<<<ga, 128, SMEM, stream>>>(Qh, Kh, Vt, Ctx, attn_out);

  // --- output projection + residual, then decomp1 ---
  k_gemm32_f32res<<<g1, 32, 0, stream>>>(Ctx, WoT, x, xr, M, D, D);
  k_decomp<<<(XN + 255) / 256, 256, 0, stream>>>(xr, x1, x1h, XN);

  // --- FFN (relu GEMM + GEMM with residual), then decomp2 -> res ---
  dim3 g2(M / 32, Dff / 32);
  k_gemm32<true, false><<<g2, 32, 0, stream>>>(x1h, W1h, Hh, M, Dff, D);
  k_gemm32_f32res<<<g1, 32, 0, stream>>>(Hh, W2h, x1, z, M, D, Dff);
  k_decomp<<<(XN + 255) / 256, 256, 0, stream>>>(z, res_out, (_Float16*)nullptr, XN);
}
